// DSCNPair_884763263408
// MI455X (gfx1250) — matrix operand order repaired and checked by reading, not compiled, not run on hardware
//
#include <hip/hip_runtime.h>
#include <hip/hip_bf16.h>
#include <math.h>

// ---------------- problem constants ----------------
#define BB   8
#define CC   256
#define HH   64
#define WW_  64
#define HWX  (HH * WW_)        // 4096
#define MTOT (BB * HWX)        // 32768
#define GG   16
#define GC_  16
#define PP   9
#define KDIM 256               // K is 256 for every GEMM in this block

typedef __bf16 bf16_t;
typedef __attribute__((ext_vector_type(16))) __bf16 v16bf;
typedef __attribute__((ext_vector_type(8)))  __bf16 v8bf;
typedef __attribute__((ext_vector_type(8)))  float  v8f_t;

// async-to-LDS builtin pointer types: (int4 AS1*, int4 AS3*, imm, imm)
typedef int v4i_t __attribute__((vector_size(16)));
typedef __attribute__((address_space(1))) v4i_t* g_v4i_p;
typedef __attribute__((address_space(3))) v4i_t* l_v4i_p;

// ============================================================
// Weight cast (+optional transpose to [N,K]) to bf16
// ============================================================
__global__ __launch_bounds__(256)
void wcast_kernel(const float* __restrict__ w, bf16_t* __restrict__ o,
                  int K, int N, int do_transpose) {
    int t = blockIdx.x * 256 + threadIdx.x;
    if (t >= K * N) return;
    if (do_transpose) {
        int n = t / K;
        int k = t - n * K;
        o[t] = (bf16_t)w[(size_t)k * N + n];   // o[n*K + k] = w[k, n]
    } else {
        o[t] = (bf16_t)w[t];
    }
}

// ============================================================
// 5x5 depthwise conv, NCHW in -> NHWC out (f32 + bf16 copy)
// ============================================================
__global__ __launch_bounds__(256)
void conv0_dw5_kernel(const float* __restrict__ x, const float* __restrict__ w,
                      const float* __restrict__ b, float* __restrict__ inp,
                      bf16_t* __restrict__ inp_bf) {
    int t   = blockIdx.x * 256 + threadIdx.x;   // MTOT*CC threads
    int c   = t & 255;
    int pos = t >> 8;
    int wo  = pos & 63;
    int ho  = (pos >> 6) & 63;
    int n   = pos >> 12;
    const float* xc = x + ((size_t)(n * CC + c)) * HWX;
    const float* wc = w + c * 25;
    float s = b[c];
#pragma unroll
    for (int ky = 0; ky < 5; ky++) {
        int hy = ho + ky - 2;
        if (hy < 0 || hy >= HH) continue;
#pragma unroll
        for (int kx = 0; kx < 5; kx++) {
            int hx = wo + kx - 2;
            if (hx < 0 || hx >= WW_) continue;
            s += xc[hy * WW_ + hx] * wc[ky * 5 + kx];
        }
    }
    size_t o = (size_t)pos * CC + c;
    inp[o]    = s;
    inp_bf[o] = (bf16_t)s;
}

// ============================================================
// 3x3 depthwise conv (NHWC) + LayerNorm over C + exact GELU -> bf16
// one block (256 threads) per spatial position
// ============================================================
__global__ __launch_bounds__(256)
void dw_ln_gelu_kernel(const float* __restrict__ inp, const float* __restrict__ w,
                       const float* __restrict__ b, const float* __restrict__ lng,
                       const float* __restrict__ lnb, bf16_t* __restrict__ x1) {
    __shared__ float s_sum[256];
    __shared__ float s_sq[256];
    int pos = blockIdx.x;
    int c   = threadIdx.x;
    int wo  = pos & 63;
    int ho  = (pos >> 6) & 63;
    int n   = pos >> 12;
    const float* wc = w + c * 9;
    float s = b[c];
#pragma unroll
    for (int dy = 0; dy < 3; dy++) {
        int hy = ho + dy - 1;
        if (hy < 0 || hy >= HH) continue;
#pragma unroll
        for (int dx = 0; dx < 3; dx++) {
            int hx = wo + dx - 1;
            if (hx < 0 || hx >= WW_) continue;
            s += inp[((size_t)(n * HWX + hy * WW_ + hx)) * CC + c] * wc[dy * 3 + dx];
        }
    }
    s_sum[c] = s;
    s_sq[c]  = s * s;
    __syncthreads();
    for (int st = 128; st > 0; st >>= 1) {
        if (c < st) { s_sum[c] += s_sum[c + st]; s_sq[c] += s_sq[c + st]; }
        __syncthreads();
    }
    float mu  = s_sum[0] * (1.0f / 256.0f);
    float var = s_sq[0] * (1.0f / 256.0f) - mu * mu;
    float v   = (s - mu) * rsqrtf(var + 1e-5f) * lng[c] + lnb[c];
    float g   = 0.5f * v * (1.0f + erff(v * 0.70710678118654752f));
    x1[(size_t)pos * CC + c] = (bf16_t)g;
}

// ============================================================
// bf16 WMMA GEMM:  out[M,N] = A[M,K] @ Bt[N,K]^T + bias[N]
// B tile (16 x K = 8KB) staged once into LDS via async-to-LDS copy,
// WMMA B fragments then come from LDS (ds_load), A streams from global.
// MODE 0: f32 out      MODE 1: bf16 out
// MODE 2: attn = gemm; out NCHW = u * attn   (pw epilogue)
// 256 threads = 8 waves; wave -> 16x16 tile; block -> 128x16
// ============================================================
template <int MODE>
__global__ __launch_bounds__(256)
void gemm_wmma_kernel(const bf16_t* __restrict__ A, const bf16_t* __restrict__ Bt,
                      const float* __restrict__ bias, void* __restrict__ outp,
                      const float* __restrict__ u, int M, int N) {
    constexpr int K = KDIM;
    __shared__ __align__(32) bf16_t sB[16 * KDIM];   // 8 KB weight tile

    const int lane   = threadIdx.x & 31;
    const int wave   = threadIdx.x >> 5;
    const int ntiles = N >> 4;
    const int bm     = blockIdx.x / ntiles;
    const int nt     = blockIdx.x - bm * ntiles;
    const int mBase  = (bm * 8 + wave) << 4;
    const int nBase  = nt << 4;

    // ---- stage B tile: 16 rows x K, contiguous 8KB at Bt + nBase*K ----
    {
        const bf16_t* gsrc = Bt + (size_t)nBase * K;
#if defined(__has_builtin) && __has_builtin(__builtin_amdgcn_global_load_async_to_lds_b128)
        for (int idx = threadIdx.x; idx < (16 * K) / 8; idx += 256) {
            __builtin_amdgcn_global_load_async_to_lds_b128(
                (g_v4i_p)(uintptr_t)(gsrc + idx * 8),
                (l_v4i_p)(uintptr_t)&sB[idx * 8],
                0, 0);
        }
#if defined(__has_builtin) && __has_builtin(__builtin_amdgcn_s_wait_asynccnt)
        __builtin_amdgcn_s_wait_asynccnt(0);
#else
        asm volatile("s_wait_asynccnt 0x0" ::: "memory");
#endif
#else
        for (int idx = threadIdx.x; idx < (16 * K) / 8; idx += 256) {
            *reinterpret_cast<v8bf*>(&sB[idx * 8]) =
                *reinterpret_cast<const v8bf*>(gsrc + idx * 8);
        }
#endif
        __syncthreads();
    }

    const int row   = mBase + (lane & 15);
    const int kb    = (lane >> 4) << 4;   // A K-subblock: 0 or 16
    const int ks    = (lane >> 4) << 4;   // B K-subblock: 0 or 16
    const int col16 = lane & 15;
    const int col   = nBase + col16;

    const bf16_t* aptr  = A + (size_t)row * K;
    const bf16_t* sbrow = &sB[col16 * K];

    v8f_t c;
#pragma unroll
    for (int i = 0; i < 8; i++) c[i] = 0.0f;

#pragma unroll
    for (int kt = 0; kt < K; kt += 32) {
        // A fragment: lane<16 -> K {kt+0..15}; lane>=16 -> {kt+16..31}
        const v8bf* pa = reinterpret_cast<const v8bf*>(aptr + kt + kb);
        v8bf alo = pa[0];
        v8bf ahi = pa[1];
        v16bf a;
#pragma unroll
        for (int i = 0; i < 8; i++) { a[i] = alo[i]; a[8 + i] = ahi[i]; }
        // B fragment from LDS: contiguous 16 K-values at lane's column
        v16bf bfrag = *reinterpret_cast<const v16bf*>(sbrow + kt + ks);
        if (kt + 32 < K) {
            __builtin_prefetch(aptr + kt + 32 + kb, 0, 3);
        }
        c = __builtin_amdgcn_wmma_f32_16x16x32_bf16(
                false, a, false, bfrag, (short)0, c, false, false);
    }

    const float bv = bias[col];
    const int   mo = (lane >> 4) << 3;   // M offset 0 or 8
    if (MODE == 0) {
        float* out = (float*)outp;
#pragma unroll
        for (int r = 0; r < 8; r++)
            out[(size_t)(mBase + mo + r) * N + col] = c[r] + bv;
    } else if (MODE == 1) {
        bf16_t* out = (bf16_t*)outp;
#pragma unroll
        for (int r = 0; r < 8; r++)
            out[(size_t)(mBase + mo + r) * N + col] = (bf16_t)(c[r] + bv);
    } else {
        float* out = (float*)outp;
#pragma unroll
        for (int r = 0; r < 8; r++) {
            int    m  = mBase + mo + r;        // spatial index n*4096 + hw
            int    nb = m >> 12;
            int    hw = m & 4095;
            size_t oi = ((size_t)nb * CC + col) * HWX + hw;  // NCHW
            out[oi] = u[oi] * (c[r] + bv);
        }
    }
}

// ============================================================
// Deformable sampling core: per (position, group) thread.
// softmax over 9 points, 9x bilinear gathers of 16 channels.
// ============================================================
__device__ __forceinline__ void gather_acc(const float* __restrict__ xproj,
                                           int n, int yu, int xu, int gbase,
                                           float w, float* acc) {
    if (yu < 0 || yu >= HH || xu < 0 || xu >= WW_) return;
    const float4* s4 = reinterpret_cast<const float4*>(
        xproj + ((size_t)(n * HWX + yu * WW_ + xu)) * CC + gbase);
#pragma unroll
    for (int j = 0; j < 4; j++) {
        float4 v = s4[j];
        acc[4 * j + 0] += w * v.x;
        acc[4 * j + 1] += w * v.y;
        acc[4 * j + 2] += w * v.z;
        acc[4 * j + 3] += w * v.w;
    }
}

__global__ __launch_bounds__(256)
void dcn_sample_kernel(const float* __restrict__ xproj, const float* __restrict__ off,
                       const float* __restrict__ mlog, bf16_t* __restrict__ ydcn) {
    int t   = blockIdx.x * 256 + threadIdx.x;   // MTOT*GG threads
    int g   = t & 15;
    int pos = t >> 4;
    int wo  = pos & 63;
    int ho  = (pos >> 6) & 63;
    int n   = pos >> 12;

    // softmax over P=9 for this group
    const float* ml = mlog + (size_t)pos * (GG * PP) + g * PP;
    float lg[PP], mx = -1e30f;
#pragma unroll
    for (int p = 0; p < PP; p++) { lg[p] = ml[p]; mx = fmaxf(mx, lg[p]); }
    float den = 0.0f;
#pragma unroll
    for (int p = 0; p < PP; p++) { lg[p] = __expf(lg[p] - mx); den += lg[p]; }
    float inv = 1.0f / den;

    const float* op = off + (size_t)pos * (GG * PP * 2) + g * (PP * 2);
    float acc[GC_];
#pragma unroll
    for (int j = 0; j < GC_; j++) acc[j] = 0.0f;
    const int gbase = g * GC_;

#pragma unroll
    for (int p = 0; p < PP; p++) {
        float wm = lg[p] * inv;
        float ox = op[2 * p + 0];
        float oy = op[2 * p + 1];
        // padded-space sampling coords: ix = wo + 1 + px + ox  (px kw-major)
        float ix = (float)wo + 1.0f + (float)(p / 3 - 1) + ox;
        float iy = (float)ho + 1.0f + (float)(p % 3 - 1) + oy;
        float xf = floorf(ix), yf = floorf(iy);
        int   x0 = (int)xf, y0 = (int)yf;
        float wx1 = ix - xf, wx0 = 1.0f - wx1;
        float wy1 = iy - yf, wy0 = 1.0f - wy1;
        // corners in padded coords -> unpadded = -1 (PAD=1); zero outside
        gather_acc(xproj, n, y0 - 1, x0 - 1, gbase, wm * wy0 * wx0, acc);
        gather_acc(xproj, n, y0 - 1, x0,     gbase, wm * wy0 * wx1, acc);
        gather_acc(xproj, n, y0,     x0 - 1, gbase, wm * wy1 * wx0, acc);
        gather_acc(xproj, n, y0,     x0,     gbase, wm * wy1 * wx1, acc);
    }

    bf16_t* dst = ydcn + (size_t)pos * CC + gbase;
#pragma unroll
    for (int j = 0; j < GC_; j++) dst[j] = (bf16_t)acc[j];
}

// ============================================================
// host-side launch
// ============================================================
extern "C" void kernel_launch(void* const* d_in, const int* in_sizes, int n_in,
                              void* d_out, int out_size, void* d_ws, size_t ws_size,
                              hipStream_t stream) {
    (void)in_sizes; (void)n_in; (void)out_size; (void)ws_size;
    const float* x       = (const float*)d_in[0];
    const float* conv0_w = (const float*)d_in[1];
    const float* conv0_b = (const float*)d_in[2];
    const float* dw_w    = (const float*)d_in[3];
    const float* dw_b    = (const float*)d_in[4];
    const float* ln_g    = (const float*)d_in[5];
    const float* ln_b    = (const float*)d_in[6];
    const float* off_w   = (const float*)d_in[7];
    const float* off_b   = (const float*)d_in[8];
    const float* msk_w   = (const float*)d_in[9];
    const float* msk_b   = (const float*)d_in[10];
    const float* in_w    = (const float*)d_in[11];
    const float* in_b    = (const float*)d_in[12];
    const float* outp_w  = (const float*)d_in[13];
    const float* outp_b  = (const float*)d_in[14];
    const float* pw_w    = (const float*)d_in[15];
    const float* pw_b    = (const float*)d_in[16];

    uint8_t* ws = (uint8_t*)d_ws;
    // aliased workspace layout (peak ~119 MB)
    const size_t SZ_F32C = (size_t)MTOT * CC * 4;          // 33.5 MB
    const size_t SZ_BF   = (size_t)MTOT * CC * 2;          // 16.8 MB
    const size_t SZ_OFF  = (size_t)MTOT * GG * PP * 2 * 4; // 37.7 MB
    const size_t SZ_MLG  = (size_t)MTOT * GG * PP * 4;     // 18.9 MB

    float*  inp_f32 = (float*)(ws + 0);                    // later reused as xproj
    float*  xproj   = inp_f32;
    bf16_t* inp_bf  = (bf16_t*)(ws + SZ_F32C);             // later reused as y bf16
    bf16_t* y_bf    = inp_bf;
    bf16_t* x1_bf   = (bf16_t*)(ws + SZ_F32C + SZ_BF);     // later reused as ydcn
    bf16_t* ydcn    = x1_bf;
    float*  offbuf  = (float*)(ws + SZ_F32C + 2 * SZ_BF);
    float*  mlog    = (float*)(ws + SZ_F32C + 2 * SZ_BF + SZ_OFF);
    uint8_t* wsw    = ws + SZ_F32C + 2 * SZ_BF + SZ_OFF + SZ_MLG;
    bf16_t* inT  = (bf16_t*)(wsw);                             // 256x256
    bf16_t* offT = (bf16_t*)(wsw + 131072);                    // 288x256
    bf16_t* mskT = (bf16_t*)(wsw + 131072 + 147456);           // 144x256
    bf16_t* outT = (bf16_t*)(wsw + 131072 + 147456 + 73728);   // 256x256
    bf16_t* pwB  = (bf16_t*)(wsw + 131072 + 147456 + 73728 + 131072);

    // 1. weight prep (bf16, [N,K])
    wcast_kernel<<<(256 * 256 + 255) / 256, 256, 0, stream>>>(in_w,   inT,  256, 256, 1);
    wcast_kernel<<<(288 * 256 + 255) / 256, 256, 0, stream>>>(off_w,  offT, 256, 288, 1);
    wcast_kernel<<<(144 * 256 + 255) / 256, 256, 0, stream>>>(msk_w,  mskT, 256, 144, 1);
    wcast_kernel<<<(256 * 256 + 255) / 256, 256, 0, stream>>>(outp_w, outT, 256, 256, 1);
    wcast_kernel<<<(256 * 256 + 255) / 256, 256, 0, stream>>>(pw_w,   pwB,  256, 256, 0); // already [N,K]

    // 2. 5x5 depthwise -> NHWC (f32 + bf16)
    conv0_dw5_kernel<<<(MTOT * CC) / 256, 256, 0, stream>>>(x, conv0_w, conv0_b,
                                                            inp_f32, inp_bf);
    // 3. 3x3 depthwise + LN + GELU -> x1 bf16
    dw_ln_gelu_kernel<<<MTOT, 256, 0, stream>>>(inp_f32, dw_w, dw_b, ln_g, ln_b, x1_bf);

    // 4. WMMA GEMMs (xproj overwrites inp_f32; reads only inp_bf)
    gemm_wmma_kernel<0><<<(MTOT / 128) * (256 / 16), 256, 0, stream>>>(
        inp_bf, inT, in_b, (void*)xproj, nullptr, MTOT, 256);
    gemm_wmma_kernel<0><<<(MTOT / 128) * (288 / 16), 256, 0, stream>>>(
        x1_bf, offT, off_b, (void*)offbuf, nullptr, MTOT, 288);
    gemm_wmma_kernel<0><<<(MTOT / 128) * (144 / 16), 256, 0, stream>>>(
        x1_bf, mskT, msk_b, (void*)mlog, nullptr, MTOT, 144);

    // 5. deformable sampling (ydcn overwrites x1 after x1's last use)
    dcn_sample_kernel<<<(MTOT * GG) / 256, 256, 0, stream>>>(xproj, offbuf, mlog, ydcn);

    // 6. output_proj -> bf16 (reuses inp_bf region)
    gemm_wmma_kernel<1><<<(MTOT / 128) * (256 / 16), 256, 0, stream>>>(
        ydcn, outT, outp_b, (void*)y_bf, nullptr, MTOT, 256);

    // 7. pw GEMM fused with out = u * attn, NCHW scatter
    gemm_wmma_kernel<2><<<(MTOT / 128) * (256 / 16), 256, 0, stream>>>(
        y_bf, pwB, pw_b, d_out, x, MTOT, 256);
}
